// OrkyMoELayer_19258633356064
// MI455X (gfx1250) — matrix operand order repaired and checked by reading, not compiled, not run on hardware
//
#include <hip/hip_runtime.h>
#include <hip/hip_bf16.h>

// ---------------------------------------------------------------------------
// MoE (top-2 of 8 experts), B=4 S=2048 D=1024 E=8 H=2048.
// Gather-based grouped GEMM on the CDNA5 WMMA pipe (v_wmma_f32_16x16x32_bf16),
// bf16 pre-converted/pre-transposed operands, double-buffered LDS pipeline.
// ---------------------------------------------------------------------------

#define NTOK 8192      // B*S
#define DDIM 1024
#define NEXP 8
#define HDIM 2048
#define TOPK 2
#define NKROWS (NTOK * TOPK)   // 16384 compacted (token,slot) rows

#define TM 128
#define TN 128
#define TK 32
#define LDA 40                 // TK + 8 bf16 pad; row stride 80B (16B aligned)

typedef __attribute__((ext_vector_type(16))) __bf16 v16bf;
typedef __attribute__((ext_vector_type(8)))  float  v8f;
typedef __attribute__((ext_vector_type(4)))  unsigned int v4u;

union FragBF { v4u u[2]; v16bf v; };

__device__ __forceinline__ unsigned short f32_to_bf16(float f) {
  union { float f; unsigned u; } x;
  x.f = f;
  unsigned r = x.u + 0x7FFFu + ((x.u >> 16) & 1u);   // round-to-nearest-even
  return (unsigned short)(r >> 16);
}

// ---------------------------------------------------------------------------
// Utility zero kernels
// ---------------------------------------------------------------------------
__global__ void zero_f32_kernel(float* __restrict__ p, int n) {
  int i = blockIdx.x * blockDim.x + threadIdx.x;
  if (i < n) p[i] = 0.0f;
}
__global__ void zero_i32_kernel(int* __restrict__ p, int n) {
  int i = blockIdx.x * blockDim.x + threadIdx.x;
  if (i < n) p[i] = 0;
}

// ---------------------------------------------------------------------------
// Elementwise f32 -> bf16 (activations)
// ---------------------------------------------------------------------------
__global__ void conv_bf16_kernel(const float* __restrict__ src,
                                 unsigned short* __restrict__ dst, int n) {
  int i = blockIdx.x * blockDim.x + threadIdx.x;
  if (i < n) dst[i] = f32_to_bf16(src[i]);
}

// ---------------------------------------------------------------------------
// Per-expert tiled transpose + convert: src [E][Kd][Nd] f32 -> dst [E][Nd][Kd]
// bf16. 32x32 tiles via LDS so both global sides stay coalesced.
// ---------------------------------------------------------------------------
__global__ __launch_bounds__(256)
void transpose_bf16_kernel(const float* __restrict__ src,
                           unsigned short* __restrict__ dst, int Kd, int Nd) {
  __shared__ float t[32][33];
  const int e = blockIdx.z;
  src += (size_t)e * Kd * Nd;
  dst += (size_t)e * Nd * Kd;
  const int n0 = blockIdx.x * 32;
  const int k0 = blockIdx.y * 32;
  const int tx = threadIdx.x & 31;
  const int ty = threadIdx.x >> 5;       // 0..7
#pragma unroll
  for (int i = 0; i < 4; ++i)
    t[ty + 8 * i][tx] = src[(size_t)(k0 + ty + 8 * i) * Nd + n0 + tx];
  __syncthreads();
#pragma unroll
  for (int i = 0; i < 4; ++i)
    dst[(size_t)(n0 + ty + 8 * i) * Kd + k0 + tx] = f32_to_bf16(t[tx][ty + 8 * i]);
}

// ---------------------------------------------------------------------------
// Router: one wave32 per token. logits = x.Wr + br ; softmax ; top-2 ; renorm
// ---------------------------------------------------------------------------
__global__ __launch_bounds__(256)
void router_kernel(const float* __restrict__ x, const float* __restrict__ Wr,
                   const float* __restrict__ br, int* __restrict__ topk_e,
                   float* __restrict__ topk_w, int* __restrict__ counts) {
  __shared__ float WrL[DDIM * NEXP];           // 32 KB of the 320 KB WGP LDS
  const int tid = threadIdx.x;
  for (int i = tid; i < DDIM * NEXP; i += 256) WrL[i] = Wr[i];
  __syncthreads();

  const int lane = tid & 31;
  const int wave = tid >> 5;
  const int tok  = blockIdx.x * 8 + wave;

  float acc[NEXP];
#pragma unroll
  for (int e = 0; e < NEXP; ++e) acc[e] = 0.0f;

  const float* xr = x + (size_t)tok * DDIM;
  for (int d = lane; d < DDIM; d += 32) {
    const float xv = xr[d];
    const float* w = WrL + d * NEXP;
#pragma unroll
    for (int e = 0; e < NEXP; ++e) acc[e] += xv * w[e];
  }
#pragma unroll
  for (int off = 16; off >= 1; off >>= 1) {
#pragma unroll
    for (int e = 0; e < NEXP; ++e) acc[e] += __shfl_xor(acc[e], off, 32);
  }

  if (lane == 0) {
    float l[NEXP];
    float m = -1e30f;
#pragma unroll
    for (int e = 0; e < NEXP; ++e) { l[e] = acc[e] + br[e]; m = fmaxf(m, l[e]); }
    float p[NEXP];
#pragma unroll
    for (int e = 0; e < NEXP; ++e) p[e] = __expf(l[e] - m);
    int a = 0;
#pragma unroll
    for (int e = 1; e < NEXP; ++e) if (p[e] > p[a]) a = e;
    int b = (a == 0) ? 1 : 0;
#pragma unroll
    for (int e = 0; e < NEXP; ++e) if (e != a && p[e] > p[b]) b = e;
    const float inv = 1.0f / (p[a] + p[b]);    // softmax denom cancels in renorm
    topk_e[tok * 2 + 0] = a;  topk_w[tok * 2 + 0] = p[a] * inv;
    topk_e[tok * 2 + 1] = b;  topk_w[tok * 2 + 1] = p[b] * inv;
    atomicAdd(&counts[a], 1);
    atomicAdd(&counts[b], 1);
  }
}

// ---------------------------------------------------------------------------
// Tiny exclusive scan over 8 expert counts + reset cursors
// ---------------------------------------------------------------------------
__global__ void scan_kernel(const int* __restrict__ counts,
                            int* __restrict__ offsets, int* __restrict__ cursors) {
  if (threadIdx.x == 0) {
    int run = 0;
    for (int e = 0; e < NEXP; ++e) {
      offsets[e] = run;
      run += counts[e];
      cursors[e] = 0;
    }
    offsets[NEXP] = run;
  }
}

// ---------------------------------------------------------------------------
// Compact (token,slot) pairs into contiguous per-expert row lists
// ---------------------------------------------------------------------------
__global__ void compact_kernel(const int* __restrict__ topk_e,
                               const float* __restrict__ topk_w,
                               const int* __restrict__ offsets,
                               int* __restrict__ cursors,
                               int* __restrict__ tok_idx, float* __restrict__ gate) {
  int i = blockIdx.x * blockDim.x + threadIdx.x;
  if (i >= NKROWS) return;
  const int e   = topk_e[i];
  const int pos = offsets[e] + atomicAdd(&cursors[e], 1);
  tok_idx[pos] = i >> 1;
  gate[pos]    = topk_w[i];
}

// ---------------------------------------------------------------------------
// Grouped GEMM on the WMMA pipe (bf16 A source + bf16 pre-transposed weights).
//   MODE 0: Hbuf = gather(Xbf).W1[e] + b1[e]
//   MODE 1: Abuf = silu(Hbuf.W2[e] + b2[e])
//   MODE 2: out += gate * (Abuf.W3[e] + b3[e])     (atomic f32 scatter)
// Block: 256 threads = 8 wave32s (4x2); tile 128x128x32; each wave: 32x64 =
// 2x4 fragments of v_wmma_f32_16x16x32_bf16. Double-buffered LDS + register
// staging: global fetch of tile k+1 issues before the WMMAs of tile k.
// ---------------------------------------------------------------------------
template <int MODE>
__global__ __launch_bounds__(256)
void moe_gemm_kernel(const unsigned short* __restrict__ Asrc, // bf16 rows
                     const unsigned short* __restrict__ WT,   // [E][Nd][Kd] bf16
                     const float* __restrict__ bias,          // [E, Ndim]
                     const int* __restrict__ offsets,         // [E+1]
                     const int* __restrict__ tok_idx,         // [NKROWS]
                     const float* __restrict__ gate,          // [NKROWS]
                     unsigned short* __restrict__ outBf,      // MODE 0/1
                     float* __restrict__ outF,                // MODE 2
                     int Kdim, int Ndim) {
  __shared__ __align__(16) unsigned short As[2][TM * LDA];  // 2 x 10 KB
  __shared__ __align__(16) unsigned short Bs[2][TN * LDA];  // 2 x 10 KB

  const int e       = blockIdx.z;
  const int rowBase = offsets[e];
  const int cnt     = offsets[e + 1] - rowBase;
  const int tileM   = blockIdx.x * TM;
  if (tileM >= cnt) return;                 // block-uniform: EXEC stays all-1s
  const int tileN   = blockIdx.y * TN;

  const int tid  = threadIdx.x;
  const int lane = tid & 31;
  const int wave = tid >> 5;
  const int wm   = wave >> 1;               // 0..3 -> M offset 32*wm
  const int wn   = wave & 1;                // 0..1 -> N offset 64*wn
  const int l16  = lane & 15;
  const bool hi  = lane >= 16;

  v8f acc[2][4];
#pragma unroll
  for (int mf = 0; mf < 2; ++mf)
#pragma unroll
    for (int nf = 0; nf < 4; ++nf)
#pragma unroll
      for (int j = 0; j < 8; ++j) acc[mf][nf][j] = 0.0f;

  // Per-thread tile-fill coordinates (hoisted row pointers)
  const int ar = tid >> 1;                  // A fill: row 0..127
  const int ak = (tid & 1) * 16;            //         k-half 0/16
  const int bn = tid >> 1;                  // B fill: col 0..127
  const int bk = (tid & 1) * 16;

  const unsigned short* aRow = nullptr;
  {
    const int grow = tileM + ar;
    if (grow < cnt) {
      const int r = (MODE == 0) ? tok_idx[rowBase + grow] : (rowBase + grow);
      aRow = Asrc + (size_t)r * Kdim;
    }
  }
  const unsigned short* bRow =
      WT + (size_t)e * Ndim * Kdim + (size_t)(tileN + bn) * Kdim;

  // Prologue: stage tile 0 in registers
  v4u a0 = {0, 0, 0, 0}, a1 = {0, 0, 0, 0}, b0, b1;
  if (aRow) {
    a0 = *(const v4u*)(aRow + ak);
    a1 = *(const v4u*)(aRow + ak + 8);
  }
  b0 = *(const v4u*)(bRow + bk);
  b1 = *(const v4u*)(bRow + bk + 8);

  const int nk = Kdim / TK;
  for (int it = 0; it < nk; ++it) {
    const int buf = it & 1;
    // commit staged registers to this iteration's LDS buffer
    *(v4u*)(&As[buf][ar * LDA + ak])     = a0;
    *(v4u*)(&As[buf][ar * LDA + ak + 8]) = a1;
    *(v4u*)(&Bs[buf][bn * LDA + bk])     = b0;
    *(v4u*)(&Bs[buf][bn * LDA + bk + 8]) = b1;
    __syncthreads();

    // issue next tile's global fetch before doing the math (latency hiding)
    if (it + 1 < nk) {
      const int k0 = (it + 1) * TK;
      if (aRow) {
        a0 = *(const v4u*)(aRow + k0 + ak);
        a1 = *(const v4u*)(aRow + k0 + ak + 8);
      }
      b0 = *(const v4u*)(bRow + k0 + bk);
      b1 = *(const v4u*)(bRow + k0 + bk + 8);
      if (it + 2 < nk) {                       // stream hint two tiles ahead
        __builtin_prefetch(bRow + (it + 2) * TK + bk, 0, 1);
      }
    }

    // ---- fragment loads per ISA 16-bit layouts + 8 WMMAs ----
    FragBF a[2], b[4];
    const int akb = hi ? 8 : 0;    // A: K halves {0..7,16..23} / {8..15,24..31}
#pragma unroll
    for (int mf = 0; mf < 2; ++mf) {
      const unsigned short* p = &As[buf][(wm * 32 + mf * 16 + l16) * LDA];
      a[mf].u[0] = *(const v4u*)(p + akb);
      a[mf].u[1] = *(const v4u*)(p + akb + 16);
    }
    const int bkb = hi ? 8 : 0;    // B: K halves {0..7,16..23} / {8..15,24..31}
#pragma unroll
    for (int nf = 0; nf < 4; ++nf) {
      const unsigned short* p = &Bs[buf][(wn * 64 + nf * 16 + l16) * LDA + bkb];
      b[nf].u[0] = *(const v4u*)(p);
      b[nf].u[1] = *(const v4u*)(p + 16);
    }
#pragma unroll
    for (int mf = 0; mf < 2; ++mf)
#pragma unroll
      for (int nf = 0; nf < 4; ++nf)
        acc[mf][nf] = __builtin_amdgcn_wmma_f32_16x16x32_bf16(
            false, a[mf].v, false, b[nf].v, (short)0, acc[mf][nf], false, false);
    // next iteration writes the other LDS buffer; the barrier above the next
    // compute keeps producers/consumers of each buffer separated.
  }

  // ---- epilogue: bias (+silu / +gate-scaled atomic scatter) ----
#pragma unroll
  for (int mf = 0; mf < 2; ++mf) {
#pragma unroll
    for (int nf = 0; nf < 4; ++nf) {
      const int col = tileN + wn * 64 + nf * 16 + l16;
      const float bv = bias[(size_t)e * Ndim + col];
#pragma unroll
      for (int j = 0; j < 8; ++j) {
        const int rloc = wm * 32 + mf * 16 + j + (hi ? 8 : 0);
        const int grow = tileM + rloc;
        if (grow < cnt) {
          float v = acc[mf][nf][j] + bv;
          if (MODE == 1) v = v * (1.0f / (1.0f + __expf(-v)));   // silu
          if (MODE == 2) {
            const int gr = rowBase + grow;
            atomicAdd(outF + (size_t)tok_idx[gr] * Ndim + col, gate[gr] * v);
          } else {
            outBf[(size_t)(rowBase + grow) * Ndim + col] = f32_to_bf16(v);
          }
        }
      }
    }
  }
}

// ---------------------------------------------------------------------------
// Host side
// ---------------------------------------------------------------------------
extern "C" void kernel_launch(void* const* d_in, const int* in_sizes, int n_in,
                              void* d_out, int out_size, void* d_ws, size_t ws_size,
                              hipStream_t stream) {
  const float* x  = (const float*)d_in[0];   // [B,S,D]
  const float* Wr = (const float*)d_in[1];   // [D,E]
  const float* br = (const float*)d_in[2];   // [E]
  const float* W1 = (const float*)d_in[3];   // [E,D,H]
  const float* b1 = (const float*)d_in[4];   // [E,H]
  const float* W2 = (const float*)d_in[5];   // [E,H,H]
  const float* b2 = (const float*)d_in[6];   // [E,H]
  const float* W3 = (const float*)d_in[7];   // [E,H,D]
  const float* b3 = (const float*)d_in[8];   // [E,D]
  float* out = (float*)d_out;                // [B,S,D] f32

  // Workspace carve-up (~272.3 MB total)
  char* ws = (char*)d_ws;
  size_t off = 0;
  auto carve = [&](size_t bytes) -> void* {
    void* p = ws + off;
    off = (off + bytes + 255) & ~(size_t)255;
    return p;
  };
  int*   counts  = (int*)carve(NEXP * sizeof(int));
  int*   cursors = (int*)carve(NEXP * sizeof(int));
  int*   offsets = (int*)carve((NEXP + 1) * sizeof(int));
  int*   topk_e  = (int*)carve(NKROWS * sizeof(int));
  float* topk_w  = (float*)carve(NKROWS * sizeof(float));
  int*   tok_idx = (int*)carve(NKROWS * sizeof(int));
  float* gate    = (float*)carve(NKROWS * sizeof(float));
  unsigned short* Xbf  = (unsigned short*)carve((size_t)NTOK * DDIM * 2);   // 16MB
  unsigned short* W1T  = (unsigned short*)carve((size_t)NEXP * DDIM * HDIM * 2); // 32MB
  unsigned short* W2T  = (unsigned short*)carve((size_t)NEXP * HDIM * HDIM * 2); // 64MB
  unsigned short* W3T  = (unsigned short*)carve((size_t)NEXP * HDIM * DDIM * 2); // 32MB
  unsigned short* Hbuf = (unsigned short*)carve((size_t)NKROWS * HDIM * 2);      // 64MB
  unsigned short* Abuf = (unsigned short*)carve((size_t)NKROWS * HDIM * 2);      // 64MB
  (void)ws_size; (void)in_sizes; (void)n_in;

  // Phase 0: clear accumulation targets
  zero_f32_kernel<<<(out_size + 255) / 256, 256, 0, stream>>>(out, out_size);
  zero_i32_kernel<<<1, 32, 0, stream>>>(counts, NEXP);

  // Phase 1: operand preparation (streaming, ~432MB total at 23.3 TB/s)
  conv_bf16_kernel<<<(NTOK * DDIM + 255) / 256, 256, 0, stream>>>(
      x, Xbf, NTOK * DDIM);
  transpose_bf16_kernel<<<dim3(HDIM / 32, DDIM / 32, NEXP), 256, 0, stream>>>(
      W1, W1T, DDIM, HDIM);
  transpose_bf16_kernel<<<dim3(HDIM / 32, HDIM / 32, NEXP), 256, 0, stream>>>(
      W2, W2T, HDIM, HDIM);
  transpose_bf16_kernel<<<dim3(DDIM / 32, HDIM / 32, NEXP), 256, 0, stream>>>(
      W3, W3T, HDIM, DDIM);

  // Phase 2: routing
  router_kernel<<<NTOK / 8, 256, 0, stream>>>(x, Wr, br, topk_e, topk_w, counts);
  scan_kernel<<<1, 32, 0, stream>>>(counts, offsets, cursors);
  compact_kernel<<<NKROWS / 256, 256, 0, stream>>>(topk_e, topk_w, offsets,
                                                   cursors, tok_idx, gate);

  // Phase 3: grouped expert GEMMs (only routed rows are computed)
  // max rows/expert = NTOK -> gx = 8192/128 = 64
  moe_gemm_kernel<0><<<dim3(64, HDIM / TN, NEXP), 256, 0, stream>>>(
      Xbf, W1T, b1, offsets, tok_idx, gate, Hbuf, nullptr, DDIM, HDIM);
  moe_gemm_kernel<1><<<dim3(64, HDIM / TN, NEXP), 256, 0, stream>>>(
      Hbuf, W2T, b2, offsets, tok_idx, gate, Abuf, nullptr, HDIM, HDIM);
  moe_gemm_kernel<2><<<dim3(64, DDIM / TN, NEXP), 256, 0, stream>>>(
      Abuf, W3T, b3, offsets, tok_idx, gate, nullptr, out, HDIM, DDIM);
}
